// GIN_43782896615722
// MI455X (gfx1250) — matrix operand order repaired and checked by reading, not compiled, not run on hardware
//
#include <hip/hip_runtime.h>

// ---------------------------------------------------------------------------
// GIN (3-layer) on MI455X / gfx1250.
//  - Edge aggregation: f32 global atomics into an L2-resident 51.2MB buffer.
//  - Node GEMMs (100000x128 @ 128x128): v_wmma_f32_16x16x32_bf16, one wave
//    per 16x16 C tile, K=128 in 4 WMMA steps. bf16 inputs / f32 accum;
//    BatchNorm directly after each GEMM renormalizes bf16 input rounding.
//  - BN stats: block partial sums + f32 atomics (tiny vs. HBM stream).
//  - Per-graph pooling fused into the 2nd BN-apply pass.
// ---------------------------------------------------------------------------

#define D 128               // D_IN == D_HID
#define D_OUT 16
#define N_LAYERS 3
#define BN_EPS 1e-5f

typedef __attribute__((ext_vector_type(8)))  __bf16 v8bf;
typedef __attribute__((ext_vector_type(16))) __bf16 v16bf;
typedef __attribute__((ext_vector_type(8)))  float  v8f;

__device__ __forceinline__ unsigned short f32_to_bf16(float f) {
  unsigned int u = __float_as_uint(f);
  u += 0x7FFFu + ((u >> 16) & 1u);      // round-to-nearest-even
  return (unsigned short)(u >> 16);
}

// ---------------- generic helpers ----------------
__global__ void k_zero(float* __restrict__ p, int n) {
  int i = blockIdx.x * blockDim.x + threadIdx.x;
  if (i < n) p[i] = 0.0f;
}

__global__ void k_copy4(float* __restrict__ dst, const float* __restrict__ src, int n4) {
  int i = blockIdx.x * blockDim.x + threadIdx.x;
  if (i < n4) ((float4*)dst)[i] = ((const float4*)src)[i];
}

__global__ void k_cvt_bf16_4(unsigned short* __restrict__ out,
                             const float* __restrict__ in, int n4) {
  int i = blockIdx.x * blockDim.x + threadIdx.x;
  if (i < n4) {
    float4 f = ((const float4*)in)[i];
    ushort4 u;
    u.x = f32_to_bf16(f.x); u.y = f32_to_bf16(f.y);
    u.z = f32_to_bf16(f.z); u.w = f32_to_bf16(f.w);
    ((ushort4*)out)[i] = u;
  }
}

// ---------------- graph bookkeeping ----------------
__global__ void k_counts(const int* __restrict__ gid, float* __restrict__ counts, int nNodes) {
  int i = blockIdx.x * blockDim.x + threadIdx.x;
  if (i < nNodes) atomicAdd(&counts[gid[i]], 1.0f);
}
__global__ void k_invert(const float* __restrict__ counts, float* __restrict__ inv, int nG) {
  int g = threadIdx.x;
  if (g < nG) inv[g] = counts[g] > 0.0f ? 1.0f / counts[g] : 0.0f;
}

// ---------------- edge scatter:  x[dst] += h[src]  ----------------
// 32 lanes per edge, float4 per lane (512B coalesced per edge row).
__global__ void k_scatter(const float* __restrict__ h, const int* __restrict__ src,
                          const int* __restrict__ dst, float* __restrict__ x, int nEdges) {
  int epb = blockDim.x >> 5;
  int e = blockIdx.x * epb + (threadIdx.x >> 5);
  if (e >= nEdges) return;
  int lane = threadIdx.x & 31;
  int s = src[e], d = dst[e];
  float4 v = *(const float4*)(h + (size_t)s * D + lane * 4);
  float* xp = x + (size_t)d * D + lane * 4;
  atomicAdd(xp + 0, v.x); atomicAdd(xp + 1, v.y);
  atomicAdd(xp + 2, v.z); atomicAdd(xp + 3, v.w);
}

// ---------------- weight convert + transpose: Wt[n][k] = bf16(W[k][n]) -----
__global__ void k_wcvt(unsigned short* __restrict__ Wt, const float* __restrict__ W) {
  int n = blockIdx.x, k = threadIdx.x;           // 128 x 128
  Wt[n * D + k] = f32_to_bf16(W[k * D + n]);
}

// ---------------- WMMA GEMM: C[M x 128] = A[M x 128] * W[128 x 128] --------
// A row-major bf16; Bt = W transposed [n][k] bf16; one wave per 16x16 tile.
__global__ void k_gemm(const unsigned short* __restrict__ A,
                       const unsigned short* __restrict__ Bt,
                       float* __restrict__ C) {
  const int mtile = blockIdx.x;                  // 6250 tiles of 16 rows
  const int wave  = threadIdx.x >> 5;            // 8 waves -> 8 n-tiles
  const int lane  = threadIdx.x & 31;
  const int m     = lane & 15;
  const int half  = lane >> 4;                   // K-half selector
  const int abase = half * 8;                    // A: K offset per ISA layout
  const int bbase = half * 8;                    // B: K offset per ISA layout
  const unsigned short* arow = A  + (size_t)(mtile * 16 + m) * D;
  const unsigned short* brow = Bt + (size_t)(wave  * 16 + m) * D;

  v8f acc = {};
#pragma unroll
  for (int k0 = 0; k0 < D; k0 += 32) {
    v8bf a_lo = *(const v8bf*)(arow + k0 + abase);        // K = k0+abase..+7
    v8bf a_hi = *(const v8bf*)(arow + k0 + 16 + abase);   // K = k0+16+abase..+7
    v8bf b_lo = *(const v8bf*)(brow + k0 + bbase);        // K = k0+bbase..+7
    v8bf b_hi = *(const v8bf*)(brow + k0 + 16 + bbase);   // K = k0+16+bbase..+7
    v16bf a = __builtin_shufflevector(a_lo, a_hi, 0,1,2,3,4,5,6,7,8,9,10,11,12,13,14,15);
    v16bf b = __builtin_shufflevector(b_lo, b_hi, 0,1,2,3,4,5,6,7,8,9,10,11,12,13,14,15);
    acc = __builtin_amdgcn_wmma_f32_16x16x32_bf16(false, a, false, b,
                                                  (short)0, acc, false, false);
  }
  // C/D layout: VGPR r -> M = r + 8*half, N = lane&15
  float* cbase = C + (size_t)(mtile * 16) * D + wave * 16 + m;
#pragma unroll
  for (int r = 0; r < 8; ++r)
    cbase[(size_t)(r + 8 * half) * D] = acc[r];
}

// ---------------- BN column statistics ----------------
#define STAT_ROWS 256
__global__ void k_stats(const float* __restrict__ y, float* __restrict__ sums,
                        float* __restrict__ sumsq, int nNodes) {
  int c  = threadIdx.x;                          // 128 threads = 128 cols
  int r0 = blockIdx.x * STAT_ROWS;
  int r1 = r0 + STAT_ROWS; if (r1 > nNodes) r1 = nNodes;
  float s = 0.0f, s2 = 0.0f;
  for (int r = r0; r < r1; ++r) {
    float v = y[(size_t)r * D + c];
    s += v; s2 += v * v;
  }
  atomicAdd(&sums[c], s);
  atomicAdd(&sumsq[c], s2);
}

__global__ void k_bnparams(const float* __restrict__ sums, const float* __restrict__ sumsq,
                           const float* __restrict__ gamma, const float* __restrict__ beta,
                           float* __restrict__ scale, float* __restrict__ shift, float invN) {
  int c = threadIdx.x;
  float mu  = sums[c]  * invN;
  float var = sumsq[c] * invN - mu * mu;
  float sc  = gamma[c] * rsqrtf(var + BN_EPS);
  scale[c] = sc;
  shift[c] = beta[c] - mu * sc;
}

// BN + ReLU + convert to bf16 (feeds GEMM2), vectorized x4
__global__ void k_bn_relu_cvt(const float* __restrict__ y, const float* __restrict__ scale,
                              const float* __restrict__ shift, unsigned short* __restrict__ out,
                              int n4) {
  int i = blockIdx.x * blockDim.x + threadIdx.x;
  if (i >= n4) return;
  int c0 = (i * 4) & (D - 1);
  float4 f = ((const float4*)y)[i];
  float vx = fmaxf(fmaf(f.x, scale[c0 + 0], shift[c0 + 0]), 0.0f);
  float vy = fmaxf(fmaf(f.y, scale[c0 + 1], shift[c0 + 1]), 0.0f);
  float vz = fmaxf(fmaf(f.z, scale[c0 + 2], shift[c0 + 2]), 0.0f);
  float vw = fmaxf(fmaf(f.w, scale[c0 + 3], shift[c0 + 3]), 0.0f);
  ushort4 u; u.x = f32_to_bf16(vx); u.y = f32_to_bf16(vy);
  u.z = f32_to_bf16(vz); u.w = f32_to_bf16(vw);
  ((ushort4*)out)[i] = u;
}

// BN + ReLU -> new h (f32) + fused per-graph pooling (pooled is L2-resident)
__global__ void k_bn_relu_pool(const float* __restrict__ z, const float* __restrict__ scale,
                               const float* __restrict__ shift, float* __restrict__ hout,
                               const int* __restrict__ gid, float* __restrict__ pooled) {
  int node = blockIdx.x;
  int c = threadIdx.x;                           // 128 threads = one node row
  float v = fmaxf(fmaf(z[(size_t)node * D + c], scale[c], shift[c]), 0.0f);
  hout[(size_t)node * D + c] = v;
  atomicAdd(&pooled[gid[node] * D + c], v);
}

// score[g][o] += (pooled[g]*inv[g]) @ Wp + bp
__global__ void k_score(const float* __restrict__ pooled, const float* __restrict__ inv,
                        const float* __restrict__ Wp, const float* __restrict__ bp,
                        float* __restrict__ out, int nG) {
  int t = blockIdx.x * blockDim.x + threadIdx.x;
  if (t >= nG * D_OUT) return;
  int g = t >> 4, o = t & (D_OUT - 1);
  float ic = inv[g];
  float s = bp[o];
  for (int k = 0; k < D; ++k)
    s += pooled[g * D + k] * ic * Wp[k * D_OUT + o];
  out[t] += s;
}

// ---------------------------------------------------------------------------
extern "C" void kernel_launch(void* const* d_in, const int* in_sizes, int n_in,
                              void* d_out, int out_size, void* d_ws, size_t ws_size,
                              hipStream_t stream) {
  const float* h_in     = (const float*)d_in[0];
  const int*   edge_src = (const int*)  d_in[1];
  const int*   edge_dst = (const int*)  d_in[2];
  const int*   graph_id = (const int*)  d_in[3];
  const float* W1       = (const float*)d_in[4];
  const float* W2       = (const float*)d_in[5];
  const float* bn1_g    = (const float*)d_in[6];
  const float* bn1_b    = (const float*)d_in[7];
  const float* bn2_g    = (const float*)d_in[8];
  const float* bn2_b    = (const float*)d_in[9];
  const float* Wp       = (const float*)d_in[10];
  const float* bp       = (const float*)d_in[11];
  float* score = (float*)d_out;

  const int nNodes = in_sizes[0] / D;            // 100000
  const int nEdges = in_sizes[1];                // 1600000
  const int nG     = out_size / D_OUT;           // 64
  const int nElem  = nNodes * D;                 // 12.8M
  const int nElem4 = nElem / 4;

  // workspace carve-up (all multiples of 256B)
  char* ws = (char*)d_ws;
  float*          hBuf   = (float*)ws;           ws += (size_t)nElem * 4;
  float*          xBuf   = (float*)ws;           ws += (size_t)nElem * 4;   // x -> y -> z
  unsigned short* bfBuf  = (unsigned short*)ws;  ws += (size_t)nElem * 2;   // bf16 activations
  unsigned short* wBuf   = (unsigned short*)ws;  ws += (size_t)D * D * 2;   // bf16 Wt
  float*          sums   = (float*)ws;           ws += 512;
  float*          sumsq  = (float*)ws;           ws += 512;
  float*          scaleB = (float*)ws;           ws += 512;
  float*          shiftB = (float*)ws;           ws += 512;
  float*          counts = (float*)ws;           ws += 512;
  float*          invc   = (float*)ws;           ws += 512;
  float*          pooled = (float*)ws;           ws += (size_t)nG * D * 4;
  (void)ws_size; (void)n_in;

  const float invN = 1.0f / (float)nNodes;
  const int mtiles = nNodes / 16;                // 6250, exact

  // init: output accumulator + graph counts
  k_zero<<<(nG * D_OUT + 255) / 256, 256, 0, stream>>>(score, nG * D_OUT);
  k_zero<<<1, 64, 0, stream>>>(counts, nG);
  k_counts<<<(nNodes + 255) / 256, 256, 0, stream>>>(graph_id, counts, nNodes);
  k_invert<<<1, 64, 0, stream>>>(counts, invc, nG);

  const float* hcur = h_in;
  for (int l = 0; l < N_LAYERS; ++l) {
    // x = h + sum_{src->dst} h[src]
    k_copy4<<<(nElem4 + 255) / 256, 256, 0, stream>>>(xBuf, hcur, nElem4);
    k_scatter<<<(nEdges + 7) / 8, 256, 0, stream>>>(hcur, edge_src, edge_dst, xBuf, nEdges);

    // GEMM1: y = bf16(x) @ W1[l]
    k_cvt_bf16_4<<<(nElem4 + 255) / 256, 256, 0, stream>>>(bfBuf, xBuf, nElem4);
    k_wcvt<<<D, D, 0, stream>>>(wBuf, W1 + (size_t)l * D * D);
    k_gemm<<<mtiles, 256, 0, stream>>>(bfBuf, wBuf, xBuf);        // y overwrites x

    // BN1 + ReLU -> bf16
    k_zero<<<1, 256, 0, stream>>>(sums, D);
    k_zero<<<1, 256, 0, stream>>>(sumsq, D);
    k_stats<<<(nNodes + STAT_ROWS - 1) / STAT_ROWS, D, 0, stream>>>(xBuf, sums, sumsq, nNodes);
    k_bnparams<<<1, D, 0, stream>>>(sums, sumsq, bn1_g + l * D, bn1_b + l * D,
                                    scaleB, shiftB, invN);
    k_bn_relu_cvt<<<(nElem4 + 255) / 256, 256, 0, stream>>>(xBuf, scaleB, shiftB, bfBuf, nElem4);

    // GEMM2: z = relu1 @ W2[l]
    k_wcvt<<<D, D, 0, stream>>>(wBuf, W2 + (size_t)l * D * D);
    k_gemm<<<mtiles, 256, 0, stream>>>(bfBuf, wBuf, xBuf);        // z overwrites y

    // BN2 + ReLU -> h_next, fused per-graph pooling
    k_zero<<<1, 256, 0, stream>>>(sums, D);
    k_zero<<<1, 256, 0, stream>>>(sumsq, D);
    k_stats<<<(nNodes + STAT_ROWS - 1) / STAT_ROWS, D, 0, stream>>>(xBuf, sums, sumsq, nNodes);
    k_bnparams<<<1, D, 0, stream>>>(sums, sumsq, bn2_g + l * D, bn2_b + l * D,
                                    scaleB, shiftB, invN);
    k_zero<<<(nG * D + 255) / 256, 256, 0, stream>>>(pooled, nG * D);
    k_bn_relu_pool<<<nNodes, D, 0, stream>>>(xBuf, scaleB, shiftB, hBuf, graph_id, pooled);

    // head: score += (pooled/count) @ Wp[l] + bp[l]
    k_score<<<(nG * D_OUT + 255) / 256, 256, 0, stream>>>(
        pooled, invc, Wp + (size_t)l * D * D_OUT, bp + (size_t)l * D_OUT, score, nG);

    hcur = hBuf;
  }
}
